// GraphColorizer_75651553951784
// MI455X (gfx1250) — matrix operand order repaired and checked by reading, not compiled, not run on hardware
//
#include <hip/hip_runtime.h>

#define N_NODES 8192
#define D_EMB   512

typedef __attribute__((ext_vector_type(16))) __bf16 v16bf;
typedef __attribute__((ext_vector_type(8)))  float  v8f;

struct Q2 { uint4 a, b; };

__device__ __forceinline__ v16bf frag_from(uint4 q0, uint4 q1) {
    Q2 t; t.a = q0; t.b = q1;
    return __builtin_bit_cast(v16bf, t);
}

// ---------------------------------------------------------------------------
// 1) Split f32 embeddings into bf16 hi/lo pair and compute row squared norms.
//    hi = truncate-to-bf16(x) (exact residual), lo = RNE-bf16(x - hi).
// ---------------------------------------------------------------------------
__global__ __launch_bounds__(256) void prep_split_kernel(
    const float* __restrict__ E,
    unsigned short* __restrict__ Ehi,
    unsigned short* __restrict__ Elo,
    float* __restrict__ sq)
{
    __shared__ float sdata[256];
    const int row = blockIdx.x;
    const size_t base = (size_t)row * D_EMB;
    float s = 0.0f;
    for (int c = threadIdx.x; c < D_EMB; c += 256) {
        const float x = E[base + c];
        const unsigned u = __float_as_uint(x);
        const unsigned short hs = (unsigned short)(u >> 16);          // truncated hi
        const float hf = __uint_as_float(u & 0xffff0000u);
        const float r  = x - hf;                                      // exact
        const unsigned ur = __float_as_uint(r);
        const unsigned short ls =
            (unsigned short)((ur + 0x7fffu + ((ur >> 16) & 1u)) >> 16); // RNE lo
        Ehi[base + c] = hs;
        Elo[base + c] = ls;
        s = fmaf(x, x, s);
    }
    sdata[threadIdx.x] = s; __syncthreads();
    for (int st = 128; st > 0; st >>= 1) {
        if (threadIdx.x < st) sdata[threadIdx.x] += sdata[threadIdx.x + st];
        __syncthreads();
    }
    if (threadIdx.x == 0) sq[row] = sdata[0];
}

// ---------------------------------------------------------------------------
// 2) Row degrees of adjacency; store reciprocal (deg >= 1 guaranteed by ring).
// ---------------------------------------------------------------------------
__global__ __launch_bounds__(256) void degree_kernel(
    const int* __restrict__ adj, float* __restrict__ rdeg)
{
    __shared__ int sdatai[256];
    const int row = blockIdx.x;
    const size_t base = (size_t)row * N_NODES;
    int s = 0;
    for (int c = threadIdx.x; c < N_NODES; c += 256) s += adj[base + c];
    sdatai[threadIdx.x] = s; __syncthreads();
    for (int st = 128; st > 0; st >>= 1) {
        if (threadIdx.x < st) sdatai[threadIdx.x] += sdatai[threadIdx.x + st];
        __syncthreads();
    }
    if (threadIdx.x == 0) rdeg[row] = 1.0f / (float)sdatai[0];
}

// ---------------------------------------------------------------------------
// 3) Column means (center). Thread-per-column keeps loads coalesced.
// ---------------------------------------------------------------------------
__global__ __launch_bounds__(256) void center_kernel(
    const float* __restrict__ E, float* __restrict__ center)
{
    const int c = blockIdx.x * 256 + threadIdx.x;
    if (c >= D_EMB) return;
    float s = 0.0f;
#pragma unroll 8
    for (int r = 0; r < N_NODES; ++r) s += E[(size_t)r * D_EMB + c];
    center[c] = s * (1.0f / (float)N_NODES);
}

// ---------------------------------------------------------------------------
// 4) Compactness: per-row ||e_i - center||, one value per row.
// ---------------------------------------------------------------------------
__global__ __launch_bounds__(256) void compact_kernel(
    const float* __restrict__ E, const float* __restrict__ center,
    float* __restrict__ partCP)
{
    __shared__ float sdata[256];
    const int row = blockIdx.x;
    const size_t base = (size_t)row * D_EMB;
    float s = 0.0f;
    for (int c = threadIdx.x; c < D_EMB; c += 256) {
        const float d = E[base + c] - center[c];
        s = fmaf(d, d, s);
    }
    sdata[threadIdx.x] = s; __syncthreads();
    for (int st = 128; st > 0; st >>= 1) {
        if (threadIdx.x < st) sdata[threadIdx.x] += sdata[threadIdx.x + st];
        __syncthreads();
    }
    if (threadIdx.x == 0) partCP[row] = sqrtf(sdata[0]);
}

// ---------------------------------------------------------------------------
// 5) Fused Gram + distance + masked loss. 128x128 block tile, 8 waves of
//    64x32 (4x2 subtiles of 16x16). 3-term bf16 split emulates f32 gram:
//    g ~= hi*hi' + hi*lo' + lo*hi', all accumulated into one f32 WMMA acc.
//    Fragment layouts follow ISA 7.12.2:
//      A (16x32 bf16): lane half h -> elems 0..7 = K+8h..K+8h+7,
//                                     elems 8..15 = K+8h+16..K+8h+23
//      B (32x16 bf16): lane half h -> same K order as A
//    (E is row-major along K, so both A and B fragments for E*E^T are
//     contiguous-K row loads: 2x b128 per fragment.)
// ---------------------------------------------------------------------------
__global__ __launch_bounds__(256) void gram_nb_loss_kernel(
    const unsigned short* __restrict__ Ehi,
    const unsigned short* __restrict__ Elo,
    const float* __restrict__ sq,
    const float* __restrict__ rdeg,
    const int* __restrict__ adj,
    float* __restrict__ partNB)
{
    __shared__ float sdata[256];
    const int tid  = threadIdx.x;
    const int wave = tid >> 5;
    const int lane = tid & 31;
    const int half = lane >> 4;
    const int l16  = lane & 15;

    const int tileM = blockIdx.y * 128;
    const int tileN = blockIdx.x * 128;
    const int waveM = tileM + (wave >> 2) * 64;   // 2 waves along M
    const int waveN = tileN + (wave & 3) * 32;    // 4 waves along N

    // Prefetch this block's 128x128 adjacency tile (64 KB) so the masked
    // epilogue hits cache after the K-loop. One 256B line per thread.
    {
        const int prow = tileM + (tid >> 1);
        const int pcol = tileN + (tid & 1) * 64;
        __builtin_prefetch(&adj[(size_t)prow * N_NODES + pcol], 0, 1);
    }

    v8f acc[4][2];
    const v8f vzero = {};
#pragma unroll
    for (int i = 0; i < 4; ++i)
#pragma unroll
        for (int j = 0; j < 2; ++j) acc[i][j] = vzero;

    for (int k0 = 0; k0 < D_EMB; k0 += 32) {
        const int kA = k0 + half * 8;
        const int kB = k0 + half * 8;
        v16bf ah[4], al[4], bh[2], bl[2];
#pragma unroll
        for (int i = 0; i < 4; ++i) {
            const size_t off = (size_t)(waveM + i * 16 + l16) * D_EMB + kA;
            const uint4* ph = (const uint4*)(Ehi + off);
            const uint4* pl = (const uint4*)(Elo + off);
            ah[i] = frag_from(ph[0], ph[2]);   // K chunks at +0 and +16 elems
            al[i] = frag_from(pl[0], pl[2]);
        }
#pragma unroll
        for (int j = 0; j < 2; ++j) {
            const size_t off = (size_t)(waveN + j * 16 + l16) * D_EMB + kB;
            const uint4* ph = (const uint4*)(Ehi + off);
            const uint4* pl = (const uint4*)(Elo + off);
            bh[j] = frag_from(ph[0], ph[2]);   // K chunks at +0 and +16 elems
            bl[j] = frag_from(pl[0], pl[2]);
        }
#pragma unroll
        for (int i = 0; i < 4; ++i)
#pragma unroll
            for (int j = 0; j < 2; ++j) {
                acc[i][j] = __builtin_amdgcn_wmma_f32_16x16x32_bf16(
                    false, ah[i], false, bh[j], (short)0, acc[i][j], false, false);
                acc[i][j] = __builtin_amdgcn_wmma_f32_16x16x32_bf16(
                    false, ah[i], false, bl[j], (short)0, acc[i][j], false, false);
                acc[i][j] = __builtin_amdgcn_wmma_f32_16x16x32_bf16(
                    false, al[i], false, bh[j], (short)0, acc[i][j], false, false);
            }
    }

    // Fused epilogue: C/D layout -> row = waveM + 16i + 8*half + r, col = waveN + 16j + l16
    float lsum = 0.0f;
#pragma unroll
    for (int i = 0; i < 4; ++i) {
#pragma unroll
        for (int r = 0; r < 8; ++r) {
            const int row  = waveM + i * 16 + half * 8 + r;
            const float sr = sq[row];
            const float rd = rdeg[row];
#pragma unroll
            for (int j = 0; j < 2; ++j) {
                const int col = waveN + j * 16 + l16;
                const float g = acc[i][j][r];
                float d2 = sr + sq[col] - 2.0f * g;
                d2 = fmaxf(d2, 1e-12f);               // clip(.,0) then max(.,1e-12)
                const float dist = sqrtf(d2);
                const int a = adj[(size_t)row * N_NODES + col];
                lsum = fmaf((float)a, dist * rd, lsum);
            }
        }
    }

    sdata[tid] = lsum; __syncthreads();
    for (int st = 128; st > 0; st >>= 1) {
        if (tid < st) sdata[tid] += sdata[tid + st];
        __syncthreads();
    }
    if (tid == 0) partNB[blockIdx.y * gridDim.x + blockIdx.x] = sdata[0];
}

// ---------------------------------------------------------------------------
// 6) Deterministic final reduction: loss = -sum(partNB) + 0.01*sum(partCP)
// ---------------------------------------------------------------------------
__global__ __launch_bounds__(256) void finalize_kernel(
    const float* __restrict__ partNB, const float* __restrict__ partCP,
    float* __restrict__ out)
{
    __shared__ float sdata[256];
    float s = 0.0f;
    for (int i = threadIdx.x; i < 64 * 64; i += 256) s += partNB[i];
    float t = 0.0f;
    for (int i = threadIdx.x; i < N_NODES; i += 256) t += partCP[i];
    sdata[threadIdx.x] = -s + 0.01f * t;
    __syncthreads();
    for (int st = 128; st > 0; st >>= 1) {
        if (threadIdx.x < st) sdata[threadIdx.x] += sdata[threadIdx.x + st];
        __syncthreads();
    }
    if (threadIdx.x == 0) out[0] = sdata[0];
}

// ---------------------------------------------------------------------------
// Launch. Workspace layout (bytes):
//   [0,        8 MiB)  Ehi   bf16 8192x512
//   [8 MiB,   16 MiB)  Elo   bf16 8192x512
//   [16 MiB, ...)      sq[8192], rdeg[8192], center[512],
//                      partNB[4096], partCP[8192]   (~114 KB)
// ---------------------------------------------------------------------------
extern "C" void kernel_launch(void* const* d_in, const int* in_sizes, int n_in,
                              void* d_out, int out_size, void* d_ws, size_t ws_size,
                              hipStream_t stream)
{
    (void)in_sizes; (void)n_in; (void)out_size; (void)ws_size;

    const float* E   = (const float*)d_in[0];   // [8192, 512] f32
    const int*   adj = (const int*)d_in[1];     // [8192, 8192] i32
    float* out = (float*)d_out;

    char* ws = (char*)d_ws;
    unsigned short* Ehi = (unsigned short*)ws;
    unsigned short* Elo = (unsigned short*)(ws + (size_t)N_NODES * D_EMB * 2);
    float* sq     = (float*)(ws + (size_t)N_NODES * D_EMB * 4);
    float* rdeg   = sq + N_NODES;
    float* center = rdeg + N_NODES;
    float* partNB = center + D_EMB;
    float* partCP = partNB + 64 * 64;

    prep_split_kernel<<<N_NODES, 256, 0, stream>>>(E, Ehi, Elo, sq);
    degree_kernel<<<N_NODES, 256, 0, stream>>>(adj, rdeg);
    center_kernel<<<(D_EMB + 255) / 256, 256, 0, stream>>>(E, center);
    compact_kernel<<<N_NODES, 256, 0, stream>>>(E, center, partCP);
    gram_nb_loss_kernel<<<dim3(64, 64), 256, 0, stream>>>(Ehi, Elo, sq, rdeg, adj, partNB);
    finalize_kernel<<<1, 256, 0, stream>>>(partNB, partCP, out);
}
